// MultiScaleAttention_79757542687128
// MI455X (gfx1250) — compile-verified, matrix operand order checked by reading, not run on hardware
//
#include <hip/hip_runtime.h>
#include <hip/hip_bf16.h>

// ---------------------------------------------------------------------------
// MultiScaleAttention for MI455X (gfx1250, wave32, WMMA).
// Pipeline: cvt->bf16, W^T pack, 3x proj GEMM (WMMA bf16), flash attention
// (WMMA QK^T + online softmax + WMMA PV), linear upsample/concat, out GEMM.
// Round 3: softmax row reductions via DPP16 row_xmask (VALU) instead of
// ds_bpermute shuffles -> no DScnt stalls in the attention inner loop.
// ---------------------------------------------------------------------------

#define D_MODEL 1024
#define N_HEADS 16
#define D_K     64
#define B_SZ    2
#define SEQ     2048
#define M_ROWS  (B_SZ * SEQ)   // 4096

typedef __bf16 bf16_t;
typedef bf16_t v16bf __attribute__((ext_vector_type(16)));
typedef float  v8f   __attribute__((ext_vector_type(8)));

__device__ __forceinline__ unsigned short f2bf(float f) {
    unsigned int u = __float_as_uint(f);
    u += 0x7FFFu + ((u >> 16) & 1u);        // round-to-nearest-even
    return (unsigned short)(u >> 16);
}

// 16-lane xor-butterfly lane swap in the VALU: DPP16 row_xmask (rows are
// 16-lane groups on wave32, so the two halves of the wave stay independent).
// dpp_ctrl = 0x160 | xor_mask.
template <int XMASK>
__device__ __forceinline__ float dpp_xor(float x) {
    int v = __builtin_amdgcn_update_dpp(0, __float_as_int(x),
                                        0x160 | XMASK, 0xf, 0xf, true);
    return __int_as_float(v);
}

__device__ __forceinline__ float rowmax16(float x) {
    x = fmaxf(x, dpp_xor<1>(x));
    x = fmaxf(x, dpp_xor<2>(x));
    x = fmaxf(x, dpp_xor<4>(x));
    x = fmaxf(x, dpp_xor<8>(x));
    return x;
}

__device__ __forceinline__ float rowsum16(float x) {
    x += dpp_xor<1>(x);
    x += dpp_xor<2>(x);
    x += dpp_xor<4>(x);
    x += dpp_xor<8>(x);
    return x;
}

// Fragment load: 16 bf16 = two 128-bit chunks (elements 0..7 and 8..15).
// Matches ISA 16-bit A/B layout: per lane, 8 contiguous K in VGPRs 0-3,
// next 8 contiguous K in VGPRs 4-7 (K base selected by lane half).
__device__ __forceinline__ v16bf load_frag(const unsigned short* lo,
                                           const unsigned short* hi) {
    v16bf f;
    ((uint4*)&f)[0] = *(const uint4*)lo;
    ((uint4*)&f)[1] = *(const uint4*)hi;
    return f;
}

__device__ __forceinline__ v8f wmma_bf16(v16bf a, v16bf b, v8f c) {
    return __builtin_amdgcn_wmma_f32_16x16x32_bf16(
        false, a, false, b, (short)0, c, false, false);
}

// ---------------------------------------------------------------------------
__global__ void cvt_f32_bf16_kernel(const float* __restrict__ in,
                                    unsigned short* __restrict__ out, int n) {
    int i = blockIdx.x * blockDim.x + threadIdx.x;
    if (i < n) out[i] = f2bf(in[i]);
}

// Wt[n][k] = bf16(W[k][n]) via LDS tile transpose (coalesced both sides).
__global__ void wtrans_kernel(const float* __restrict__ W,
                              unsigned short* __restrict__ Wt) {
    __shared__ float tile[32][33];
    int tx = threadIdx.x, ty = threadIdx.y;
    int n0 = blockIdx.x * 32, k0 = blockIdx.y * 32;
    for (int i = 0; i < 32; i += 8)
        tile[ty + i][tx] = W[(size_t)(k0 + ty + i) * D_MODEL + n0 + tx];
    __syncthreads();
    for (int i = 0; i < 32; i += 8)
        Wt[(size_t)(n0 + ty + i) * D_MODEL + k0 + tx] = f2bf(tile[tx][ty + i]);
}

// ---------------------------------------------------------------------------
// GEMM: Y[4096,1024] = X[4096,1024](bf16) * Wt^T + bias.  Wave tile 32x64.
// MODE 0: Q head-major (scaled 1/8, stride-2 downsample heads 8..15)
// MODE 1: K head-major (downsampled)
// MODE 2: V transposed per head  Vt[b][h][d][s] (downsampled)
// MODE 3: plain f32 row-major (final output projection)
template <int MODE>
__global__ __launch_bounds__(256) void gemm_bf16_kernel(
    const unsigned short* __restrict__ X,
    const unsigned short* __restrict__ Wt,
    const float* __restrict__ bias,
    float* __restrict__ outF,
    unsigned short* __restrict__ outH) {
    const int lane  = threadIdx.x & 31;
    const int wave  = threadIdx.x >> 5;
    const int idx16 = lane & 15;
    const int half  = lane >> 4;
    const int w     = blockIdx.x * 8 + wave;   // 2048 wave tiles
    const int tm    = w & 127;                 // 128 row tiles (32 rows)
    const int tn    = w >> 7;                  // 16 col tiles (64 cols)
    const int row0  = tm * 32, col0 = tn * 64;

    const v8f vzero = {0.f, 0.f, 0.f, 0.f, 0.f, 0.f, 0.f, 0.f};
    v8f acc[2][4];
#pragma unroll
    for (int a = 0; a < 2; a++)
#pragma unroll
        for (int t = 0; t < 4; t++) acc[a][t] = vzero;

    for (int kc = 0; kc < D_MODEL; kc += 32) {
        v16bf afr[2], bfr[4];
#pragma unroll
        for (int a = 0; a < 2; a++) {
            const unsigned short* base =
                X + (size_t)(row0 + a * 16 + idx16) * D_MODEL + kc + half * 8;
            afr[a] = load_frag(base, base + 16);
        }
#pragma unroll
        for (int t = 0; t < 4; t++) {
            const unsigned short* base =
                Wt + (size_t)(col0 + t * 16 + idx16) * D_MODEL + kc + half * 8;
            bfr[t] = load_frag(base, base + 16);
        }
        // gfx1250 prefetch of next K chunk (global_prefetch_b8)
        __builtin_prefetch(X + (size_t)(row0 + idx16) * D_MODEL + kc + 32, 0, 1);
#pragma unroll
        for (int a = 0; a < 2; a++)
#pragma unroll
            for (int t = 0; t < 4; t++)
                acc[a][t] = wmma_bf16(afr[a], bfr[t], acc[a][t]);
    }

    // Epilogue: C layout is VGPR j -> row j+8*half, lane%16 -> column.
#pragma unroll
    for (int a = 0; a < 2; a++)
#pragma unroll
        for (int t = 0; t < 4; t++) {
            const int c = col0 + t * 16 + idx16;     // column for this lane
            const float bc = bias[c];
#pragma unroll
            for (int j = 0; j < 8; j++) {
                const int r = row0 + a * 16 + j + 8 * half;
                float v = acc[a][t][j] + bc;
                if (MODE == 3) {
                    outF[(size_t)r * D_MODEL + c] = v;
                } else {
                    const int b = r >> 11, s = r & (SEQ - 1);
                    const int h = c >> 6,  d = c & (D_K - 1);
                    if (MODE == 0) v *= 0.125f;        // fold 1/sqrt(64)
                    int sd = s;
                    if (h >= 8) {                      // scale-2 heads
                        if (s & 1) continue;           // stride-2 sample
                        sd = s >> 1;
                    }
                    if (MODE == 2)
                        outH[((size_t)(b * N_HEADS + h) * D_K + d) * SEQ + sd] = f2bf(v);
                    else
                        outH[((size_t)(b * N_HEADS + h) * SEQ + sd) * D_K + d] = f2bf(v);
                }
            }
        }
}

// ---------------------------------------------------------------------------
// Flash attention: 1 wave per 16-query tile; 32-key chunks.
// 4 WMMA for scores + 4 WMMA for PV per chunk; probabilities re-enter the
// A-fragment layout via a 1KB per-wave LDS staging tile.  Row reductions
// are DPP16 row_xmask butterflies (pure VALU).
__global__ __launch_bounds__(128) void attn_kernel(
    const unsigned short* __restrict__ Qh,
    const unsigned short* __restrict__ Kh,
    const unsigned short* __restrict__ Vt,
    float* __restrict__ Oh) {
    __shared__ __align__(16) unsigned short plds[4][16][32];

    const int lane  = threadIdx.x & 31;
    const int wave  = threadIdx.x >> 5;
    const int idx16 = lane & 15;
    const int half  = lane >> 4;
    const int h = blockIdx.y, b = blockIdx.z;
    const int L = (h < 8) ? SEQ : (SEQ / 2);
    const int qt = blockIdx.x * 4 + wave;
    if (qt * 16 >= L) return;
    const int q0 = qt * 16;

    const unsigned short* Qb = Qh + (size_t)(b * N_HEADS + h) * SEQ * D_K;
    const unsigned short* Kb = Kh + (size_t)(b * N_HEADS + h) * SEQ * D_K;
    const unsigned short* Vb = Vt + (size_t)(b * N_HEADS + h) * D_K * SEQ;

    v16bf qa[2];
#pragma unroll
    for (int c = 0; c < 2; c++) {
        const unsigned short* base = Qb + (size_t)(q0 + idx16) * D_K + c * 32 + half * 8;
        qa[c] = load_frag(base, base + 16);
    }

    const v8f vzero = {0.f, 0.f, 0.f, 0.f, 0.f, 0.f, 0.f, 0.f};
    v8f acc[4];
#pragma unroll
    for (int t = 0; t < 4; t++) acc[t] = vzero;
    float mrow[8], lrow[8];
#pragma unroll
    for (int j = 0; j < 8; j++) { mrow[j] = -1e30f; lrow[j] = 0.0f; }

    for (int k0 = 0; k0 < L; k0 += 32) {
        // scores: 16 queries x 32 keys as two 16x16 C tiles
        v8f sc[2]; sc[0] = vzero; sc[1] = vzero;
#pragma unroll
        for (int nt = 0; nt < 2; nt++) {
            const unsigned short* krow = Kb + (size_t)(k0 + nt * 16 + idx16) * D_K;
#pragma unroll
            for (int c = 0; c < 2; c++) {
                const unsigned short* base = krow + c * 32 + half * 8;
                v16bf kb = load_frag(base, base + 16);
                sc[nt] = wmma_bf16(qa[c], kb, sc[nt]);
            }
        }
        // online softmax per row (rows live in halves; DPP16 row_xmask
        // reductions stay inside each 16-lane row on wave32)
#pragma unroll
        for (int j = 0; j < 8; j++) {
            float mx   = rowmax16(fmaxf(sc[0][j], sc[1][j]));
            float mnew = fmaxf(mrow[j], mx);
            float alpha = __expf(mrow[j] - mnew);
            float e0 = __expf(sc[0][j] - mnew);
            float e1 = __expf(sc[1][j] - mnew);
            float rs = rowsum16(e0 + e1);
            lrow[j] = lrow[j] * alpha + rs;
            mrow[j] = mnew;
#pragma unroll
            for (int t = 0; t < 4; t++) acc[t][j] *= alpha;
            int r = j + 8 * half;
            plds[wave][r][idx16]      = f2bf(e0);
            plds[wave][r][16 + idx16] = f2bf(e1);
        }
        // P back as A fragment (wave-private LDS; DScnt wait inserted by cc)
        v16bf pa = load_frag(&plds[wave][idx16][half * 8],
                             &plds[wave][idx16][16 + half * 8]);
#pragma unroll
        for (int t = 0; t < 4; t++) {
            const unsigned short* vr =
                Vb + (size_t)(t * 16 + idx16) * SEQ + k0 + half * 8;
            v16bf vb = load_frag(vr, vr + 16);
            acc[t] = wmma_bf16(pa, vb, acc[t]);
        }
    }

    float* Ob = Oh + (size_t)(b * N_HEADS + h) * SEQ * D_K;
#pragma unroll
    for (int t = 0; t < 4; t++)
#pragma unroll
        for (int j = 0; j < 8; j++) {
            int r = q0 + j + 8 * half;
            Ob[(size_t)r * D_K + t * 16 + idx16] = acc[t][j] / lrow[j];
        }
}

// ---------------------------------------------------------------------------
// Linear upsample (align_corners=False) for heads 8..15 + concat -> bf16 ctx.
__global__ void upsample_ctx_kernel(const float* __restrict__ Oh,
                                    unsigned short* __restrict__ Ctx) {
    int i = blockIdx.x * blockDim.x + threadIdx.x;
    if (i >= B_SZ * SEQ * D_MODEL) return;
    int c = i & (D_MODEL - 1);
    int s = (i >> 10) & (SEQ - 1);
    int b = i >> 21;
    int h = c >> 6, d = c & (D_K - 1);
    const float* Ob = Oh + (size_t)(b * N_HEADS + h) * SEQ * D_K;
    float v;
    if (h < 8) {
        v = Ob[(size_t)s * D_K + d];
    } else {
        float coord = (s + 0.5f) * 0.5f - 0.5f;     // l=1024 -> 2048
        coord = fmaxf(coord, 0.0f);
        int i0 = (int)coord; if (i0 > 1023) i0 = 1023;
        int i1 = i0 + 1;     if (i1 > 1023) i1 = 1023;
        float w = coord - (float)i0;
        v = Ob[(size_t)i0 * D_K + d] * (1.0f - w) + Ob[(size_t)i1 * D_K + d] * w;
    }
    Ctx[i] = f2bf(v);
}

// ---------------------------------------------------------------------------
extern "C" void kernel_launch(void* const* d_in, const int* in_sizes, int n_in,
                              void* d_out, int out_size, void* d_ws, size_t ws_size,
                              hipStream_t stream) {
    const float* q   = (const float*)d_in[0];
    const float* k   = (const float*)d_in[1];
    const float* v   = (const float*)d_in[2];
    const float* w_q = (const float*)d_in[3];
    const float* b_q = (const float*)d_in[4];
    const float* w_k = (const float*)d_in[5];
    const float* b_k = (const float*)d_in[6];
    const float* w_v = (const float*)d_in[7];
    const float* b_v = (const float*)d_in[8];
    const float* w_o = (const float*)d_in[9];
    const float* b_o = (const float*)d_in[10];
    float* out = (float*)d_out;

    char* ws = (char*)d_ws;
    const size_t SZ_X = (size_t)M_ROWS * D_MODEL * 2;   // 8 MB (bf16 4096x1024)
    const size_t SZ_W = (size_t)D_MODEL * D_MODEL * 2;  // 2 MB
    unsigned short* Xq  = (unsigned short*)(ws);
    unsigned short* Xk  = (unsigned short*)(ws + SZ_X);
    unsigned short* Xv  = (unsigned short*)(ws + 2 * SZ_X);
    unsigned short* Wqt = (unsigned short*)(ws + 3 * SZ_X);
    unsigned short* Wkt = (unsigned short*)(ws + 3 * SZ_X + SZ_W);
    unsigned short* Wvt = (unsigned short*)(ws + 3 * SZ_X + 2 * SZ_W);
    unsigned short* Wot = (unsigned short*)(ws + 3 * SZ_X + 3 * SZ_W);
    unsigned short* Qh  = (unsigned short*)(ws + 3 * SZ_X + 4 * SZ_W);
    unsigned short* Kh  = (unsigned short*)(ws + 4 * SZ_X + 4 * SZ_W);
    unsigned short* Vt  = (unsigned short*)(ws + 5 * SZ_X + 4 * SZ_W);
    float*          OhP = (float*)         (ws + 6 * SZ_X + 4 * SZ_W); // 16 MB
    unsigned short* Ctx = (unsigned short*)(ws + 8 * SZ_X + 4 * SZ_W);

    const int nX = M_ROWS * D_MODEL;
    cvt_f32_bf16_kernel<<<(nX + 255) / 256, 256, 0, stream>>>(q, Xq, nX);
    cvt_f32_bf16_kernel<<<(nX + 255) / 256, 256, 0, stream>>>(k, Xk, nX);
    cvt_f32_bf16_kernel<<<(nX + 255) / 256, 256, 0, stream>>>(v, Xv, nX);

    dim3 tb(32, 8), tg(32, 32);
    wtrans_kernel<<<tg, tb, 0, stream>>>(w_q, Wqt);
    wtrans_kernel<<<tg, tb, 0, stream>>>(w_k, Wkt);
    wtrans_kernel<<<tg, tb, 0, stream>>>(w_v, Wvt);
    wtrans_kernel<<<tg, tb, 0, stream>>>(w_o, Wot);

    gemm_bf16_kernel<0><<<256, 256, 0, stream>>>(Xq, Wqt, b_q, nullptr, Qh);
    gemm_bf16_kernel<1><<<256, 256, 0, stream>>>(Xk, Wkt, b_k, nullptr, Kh);
    gemm_bf16_kernel<2><<<256, 256, 0, stream>>>(Xv, Wvt, b_v, nullptr, Vt);

    attn_kernel<<<dim3(32, N_HEADS, B_SZ), 128, 0, stream>>>(Qh, Kh, Vt, OhP);

    upsample_ctx_kernel<<<(nX + 255) / 256, 256, 0, stream>>>(OhP, Ctx);

    gemm_bf16_kernel<3><<<256, 256, 0, stream>>>(Ctx, Wot, b_o, out, nullptr);
}
